// LSTM_21199958573341
// MI455X (gfx1250) — matrix operand order repaired and checked by reading, not compiled, not run on hardware
//
#include <hip/hip_runtime.h>
#include <hip/hip_bf16.h>

#define B_SZ  256
#define T_SZ  512
#define IN0   128
#define HID   256
#define NCLS  100

typedef __attribute__((ext_vector_type(16))) __bf16          bf16x16;
typedef __attribute__((ext_vector_type(8)))  __bf16          bf16x8;
typedef __attribute__((ext_vector_type(8)))  float           f32x8;
typedef __attribute__((ext_vector_type(8)))  unsigned short  u16x8;
typedef __attribute__((ext_vector_type(4)))  unsigned int    u32x4;
typedef __attribute__((ext_vector_type(8)))  int             i32x8;
typedef __attribute__((ext_vector_type(4)))  int             i32x4;

#ifndef USE_TDM
#if __has_builtin(__builtin_amdgcn_tensor_load_to_lds) && __has_builtin(__builtin_amdgcn_s_wait_tensorcnt)
#define USE_TDM 1
#else
#define USE_TDM 0
#endif
#endif

__device__ __forceinline__ unsigned short f2bf(float f) {
  union { float f; unsigned u; } v; v.f = f;
  unsigned r = v.u + 0x7FFFu + ((v.u >> 16) & 1u);   // round-to-nearest-even
  return (unsigned short)(r >> 16);
}
__device__ __forceinline__ float bf2f(unsigned short h) {
  union { unsigned u; float f; } v; v.u = ((unsigned)h) << 16;
  return v.f;
}
__device__ __forceinline__ float sigm(float x)  { return 1.0f / (1.0f + __expf(-x)); }
__device__ __forceinline__ float tanhf_(float x){ return 2.0f * sigm(2.0f * x) - 1.0f; }

#if USE_TDM
// Issue one 2D TDM tile load: rows x tile_w (2-byte elements), strided source rows,
// contiguous LDS rows with pad_amount(code) DWORDs inserted every pad_interval(code) DWORDs.
// D# bitfields per CDNA5 ISA ch.8 (group0: count/lds/global/type, group1: dims/strides/pad).
// This toolchain's builtin takes 6 args: (g0, g1, g2, g3, g_extra, cpol).
__device__ __forceinline__ void tdm_load_2d(unsigned lds_off, const void* gaddr,
                                            unsigned tile_w, unsigned rows,
                                            unsigned long long row_stride,
                                            unsigned pi_code, unsigned pa_code) {
  unsigned long long ga = (unsigned long long)gaddr;
  u32x4 g0;
  g0[0] = 1u;                                            // count=1, user mode
  g0[1] = lds_off;                                       // LDS byte address
  g0[2] = (unsigned)(ga & 0xffffffffu);                  // global_addr[31:0]
  g0[3] = (unsigned)((ga >> 32) & 0x1ffffffu) | (2u << 30); // global_addr[56:32] | type=2
  i32x8 g1;
  g1[0] = (int)((1u << 16) | (1u << 20) | (pi_code << 22) | (pa_code << 25)); // 2B elts, pad on
  g1[1] = (int)((tile_w & 0xffffu) << 16);               // tensor_dim0[15:0]
  g1[2] = (int)(((tile_w >> 16) & 0xffffu) | ((rows & 0xffffu) << 16)); // td0 hi | td1 lo
  g1[3] = (int)(((rows >> 16) & 0xffffu) | ((tile_w & 0xffffu) << 16)); // td1 hi | tile_dim0
  g1[4] = (int)(rows & 0xffffu);                         // tile_dim1 (tile_dim2=0)
  g1[5] = (int)(row_stride & 0xffffffffull);             // tensor_dim0_stride[31:0]
  g1[6] = (int)((row_stride >> 32) & 0xffffull);         // stride[47:32] (dim1_stride=0)
  g1[7] = 0;
  i32x4 z4 = (i32x4)0;                                   // groups 2/3 unused (2D tensor)
  i32x8 z8 = (i32x8)0;
  __builtin_amdgcn_tensor_load_to_lds(g0, g1, z4, z4, z8, 0);
}
#endif

// ---------------------------------------------------------------- prep kernels
__global__ void init_counters(unsigned int* c) {
  if (threadIdx.x < 128) c[threadIdx.x] = 0u;
}

__global__ void cvt_f32_to_bf16(const float* __restrict__ src,
                                unsigned short* __restrict__ dst, int n) {
  int i = blockIdx.x * blockDim.x + threadIdx.x;
  if (i < n) dst[i] = f2bf(src[i]);
}

// wcomb[n][k] (bf16, row-major, 1024 x K): k<IN_W -> W_ih[n][k], else W_hh[n][k-IN_W]
__global__ void pack_weights(const float* __restrict__ Wih, const float* __restrict__ Whh,
                             unsigned short* __restrict__ wcomb, int IN_W) {
  int K = IN_W + HID;
  int idx = blockIdx.x * blockDim.x + threadIdx.x;
  if (idx >= 1024 * K) return;
  int n = idx / K;
  int k = idx - n * K;
  float v = (k < IN_W) ? Wih[n * IN_W + k] : Whh[n * HID + (k - IN_W)];
  wcomb[idx] = f2bf(v);
}

// ---------------------------------------------------------- persistent LSTM layer
// grid = 32 WGs (8 batch-tiles of 32 x 4 col-groups of 64 h-cols), block = 256 (8 waves)
template <int IN_W>
__global__ void __launch_bounds__(256, 1)
lstm_layer_kernel(const unsigned short* __restrict__ in_seq,   // [B, T, IN_W] bf16
                  unsigned short* __restrict__ h_seq,          // [B, T, HID]  bf16 (out)
                  const unsigned short* __restrict__ wcomb,    // [1024, K]    bf16
                  const float* __restrict__ b_ih, const float* __restrict__ b_hh,
                  unsigned int* __restrict__ counters) {
  constexpr int K   = IN_W + HID;
  constexpr int WKP = K + 8;          // weight LDS pitch (elements)
  constexpr int XP  = IN_W + 8;       // x-panel pitch
  constexpr int HP  = HID + 8;        // h-panel pitch
  extern __shared__ unsigned short smem[];
  unsigned short* wlds  = smem;                         // 256 * WKP
  unsigned short* axlds = smem + 256 * WKP;             // 32 * XP
  unsigned short* ahlds = axlds + 32 * XP;              // 32 * HP

  const int tid   = threadIdx.x;
  const int btile = blockIdx.x >> 2;              // 0..7
  const int cgrp  = blockIdx.x & 3;               // 0..3
  unsigned int* counter = counters + btile * 16;  // per-batch-tile barrier (4 WGs)

  // -------- load resident weight slice: local col n -> global gate row
  constexpr int CPR = K >> 3;
  for (int idx = tid; idx < 256 * CPR; idx += 256) {
    int n  = idx / CPR;
    int kk = (idx - n * CPR) << 3;
    int gidx = ((n >> 6) << 8) + (cgrp << 6) + (n & 63);   // gate*256 + cgrp*64 + local
    *(u16x8*)(wlds + n * WKP + kk) = *(const u16x8*)(wcomb + (size_t)gidx * K + kk);
  }
  // zero h panel (consumed as h_{-1} = 0 at t=0)
  for (int idx = tid; idx < 32 * HP / 8; idx += 256)
    *(u16x8*)(ahlds + idx * 8) = (u16x8)0;
  __syncthreads();

  // -------- per-wave geometry (ISA WMMA VGPR layouts)
  const int lane  = tid & 31;
  const int r     = (tid >> 5) & 1;      // batch row-tile (16 rows)
  const int q     = tid >> 6;            // h col-tile (16 cols)
  const int ln    = lane & 15;
  const int hiL   = lane >> 4;
  const int koffA = hiL << 3;            // A frag: lane>=16 -> K+8
  const int koffB = hiL << 3;            // B frag: lane>=16 -> K+8
  const int mrow  = (r << 4) + ln;
  const int colh  = (q << 4) + ln;       // local h col (0..63)
  const int gcol  = (cgrp << 6) + colh;  // global h col (0..255)

  float bias[4];
#pragma unroll
  for (int g = 0; g < 4; ++g) bias[g] = b_ih[g * HID + gcol] + b_hh[g * HID + gcol];

  f32x8 cstate = (f32x8)0.0f;
  const size_t hbstride = (size_t)T_SZ * HID;

  auto loadA = [&](int k0) {
    const unsigned short* ap = (k0 < IN_W)
        ? (axlds + mrow * XP + k0 + koffA)
        : (ahlds + mrow * HP + (k0 - IN_W) + koffA);
    bf16x8 lo = *(const bf16x8*)(ap);
    bf16x8 hi = *(const bf16x8*)(ap + 16);
    return __builtin_shufflevector(lo, hi, 0,1,2,3,4,5,6,7,8,9,10,11,12,13,14,15);
  };
  auto loadB = [&](int k0, int g) {
    const unsigned short* bp = wlds + ((g << 6) + colh) * WKP + k0 + koffB;
    bf16x8 lo = *(const bf16x8*)(bp);
    bf16x8 hi = *(const bf16x8*)(bp + 16);
    return __builtin_shufflevector(lo, hi, 0,1,2,3,4,5,6,7,8,9,10,11,12,13,14,15);
  };

#if USE_TDM
  const unsigned ax_off = (unsigned)(unsigned long long)(const void*)axlds;
  const unsigned ah_off = (unsigned)(unsigned long long)(const void*)ahlds;
  constexpr unsigned PI_X = (IN_W == 128) ? 5u : 6u;   // 64 / 128 DWORD rows
#endif

  for (int t = 0; t < T_SZ; ++t) {
    // ---- stage A = [x_t | h_{t-1}] panels for our 32 batch rows
#if USE_TDM
    if (tid < 32) {                       // wave 0 issues the DMAs (EXEC ignored by TDM)
      tdm_load_2d(ax_off, in_seq + ((size_t)(btile << 5) * T_SZ + t) * IN_W,
                  (unsigned)IN_W, 32u, (unsigned long long)T_SZ * IN_W, PI_X, 3u);
      if (t > 0)
        tdm_load_2d(ah_off, h_seq + (size_t)(btile << 5) * hbstride + (size_t)(t - 1) * HID,
                    (unsigned)HID, 32u, (unsigned long long)hbstride, 6u, 3u);
      __builtin_amdgcn_s_wait_tensorcnt((unsigned short)0);
    }
#else
    for (int idx = tid; idx < 32 * (IN_W >> 3); idx += 256) {
      int row = idx / (IN_W >> 3);
      int kk  = (idx - row * (IN_W >> 3)) << 3;
      int b   = (btile << 5) + row;
      *(u16x8*)(axlds + row * XP + kk) =
          *(const u16x8*)(in_seq + ((size_t)b * T_SZ + t) * IN_W + kk);
    }
    if (t > 0) {
      for (int idx = tid; idx < 32 * (HID >> 3); idx += 256) {
        int row = idx >> 5;
        int kk  = (idx & 31) << 3;
        int b   = (btile << 5) + row;
        *(u16x8*)(ahlds + row * HP + kk) =
            *(const u16x8*)(h_seq + (size_t)b * hbstride + (size_t)(t - 1) * HID + kk);
      }
    }
#endif
    if (t + 1 < T_SZ && tid < 32) {      // prefetch next step's input rows (near)
      int b = (btile << 5) + tid;
      __builtin_prefetch(in_seq + ((size_t)b * T_SZ + (t + 1)) * IN_W, 0, 3);
    }
    __syncthreads();

    // ---- gates = bias + [x|h] @ W^T  (bf16 WMMA, fp32 accum), software-pipelined
    f32x8 acc0, acc1, acc2, acc3;
#pragma unroll
    for (int j = 0; j < 8; ++j) {
      acc0[j] = bias[0]; acc1[j] = bias[1]; acc2[j] = bias[2]; acc3[j] = bias[3];
    }
    bf16x16 aC  = loadA(0);
    bf16x16 bC0 = loadB(0, 0), bC1 = loadB(0, 1), bC2 = loadB(0, 2), bC3 = loadB(0, 3);
#pragma unroll
    for (int k0 = 0; k0 < K; k0 += 32) {
      bf16x16 aN = aC, bN0 = bC0, bN1 = bC1, bN2 = bC2, bN3 = bC3;
      if (k0 + 32 < K) {                 // prefetch next K-slice fragments from LDS
        aN  = loadA(k0 + 32);
        bN0 = loadB(k0 + 32, 0); bN1 = loadB(k0 + 32, 1);
        bN2 = loadB(k0 + 32, 2); bN3 = loadB(k0 + 32, 3);
      }
      acc0 = __builtin_amdgcn_wmma_f32_16x16x32_bf16(false, aC, false, bC0, (short)0, acc0, false, false);
      acc1 = __builtin_amdgcn_wmma_f32_16x16x32_bf16(false, aC, false, bC1, (short)0, acc1, false, false);
      acc2 = __builtin_amdgcn_wmma_f32_16x16x32_bf16(false, aC, false, bC2, (short)0, acc2, false, false);
      acc3 = __builtin_amdgcn_wmma_f32_16x16x32_bf16(false, aC, false, bC3, (short)0, acc3, false, false);
      aC = aN; bC0 = bN0; bC1 = bN1; bC2 = bN2; bC3 = bN3;
    }

    // ---- nonlinearity + cell update + h store (register-local, C/D layout)
#pragma unroll
    for (int j = 0; j < 8; ++j) {
      float iv = sigm(acc0[j]);
      float fv = sigm(acc1[j]);
      float gv = tanhf_(acc2[j]);
      float ov = sigm(acc3[j]);
      float c  = fv * cstate[j] + iv * gv;
      cstate[j] = c;
      float hv = ov * tanhf_(c);
      int bg = (btile << 5) + (r << 4) + (hiL << 3) + j;
      h_seq[(size_t)bg * hbstride + (size_t)t * HID + gcol] = f2bf(hv);
    }

    // ---- per-batch-tile barrier: only the 4 col-group WGs must agree
    __threadfence();
    __syncthreads();
    if (tid == 0) {
      __hip_atomic_fetch_add(counter, 1u, __ATOMIC_RELEASE, __HIP_MEMORY_SCOPE_AGENT);
      unsigned target = (unsigned)(t + 1) * 4u;
      while (__hip_atomic_load(counter, __ATOMIC_ACQUIRE, __HIP_MEMORY_SCOPE_AGENT) < target)
        __builtin_amdgcn_s_sleep(1);
    }
    __syncthreads();
    __threadfence();
  }
}

// ---------------------------------------------------------------- final FC
__global__ void fc_kernel(const unsigned short* __restrict__ hseq,
                          const float* __restrict__ fw, const float* __restrict__ fb,
                          float* __restrict__ out) {
  int b = blockIdx.x;
  int cls = threadIdx.x;
  if (cls >= NCLS) return;
  const unsigned short* hrow = hseq + ((size_t)b * T_SZ + (T_SZ - 1)) * HID;
  float acc = fb[cls];
  for (int k = 0; k < HID; ++k) acc += bf2f(hrow[k]) * fw[cls * HID + k];
  out[b * NCLS + cls] = acc;
}

// ---------------------------------------------------------------- launch
extern "C" void kernel_launch(void* const* d_in, const int* in_sizes, int n_in,
                              void* d_out, int out_size, void* d_ws, size_t ws_size,
                              hipStream_t stream) {
  (void)in_sizes; (void)n_in; (void)out_size; (void)ws_size;
  const float* x    = (const float*)d_in[0];
  const float* Wih0 = (const float*)d_in[1];
  const float* Whh0 = (const float*)d_in[2];
  const float* bih0 = (const float*)d_in[3];
  const float* bhh0 = (const float*)d_in[4];
  const float* Wih1 = (const float*)d_in[5];
  const float* Whh1 = (const float*)d_in[6];
  const float* bih1 = (const float*)d_in[7];
  const float* bhh1 = (const float*)d_in[8];
  const float* fcw  = (const float*)d_in[9];
  const float* fcb  = (const float*)d_in[10];
  float* out = (float*)d_out;

  char* ws = (char*)d_ws;
  size_t off = 0;
  unsigned int*   counters = (unsigned int*)(ws + off);   off += 1024;
  unsigned short* xbf      = (unsigned short*)(ws + off); off += (size_t)B_SZ * T_SZ * IN0 * 2;
  unsigned short* hseq0    = (unsigned short*)(ws + off); off += (size_t)B_SZ * T_SZ * HID * 2;
  unsigned short* hseq1    = (unsigned short*)(ws + off); off += (size_t)B_SZ * T_SZ * HID * 2;
  unsigned short* wc0      = (unsigned short*)(ws + off); off += (size_t)1024 * (IN0 + HID) * 2;
  unsigned short* wc1      = (unsigned short*)(ws + off); off += (size_t)1024 * (HID + HID) * 2;

  init_counters<<<1, 128, 0, stream>>>(counters);
  { int n = B_SZ * T_SZ * IN0;  cvt_f32_to_bf16<<<(n + 255) / 256, 256, 0, stream>>>(x, xbf, n); }
  { int n = 1024 * (IN0 + HID); pack_weights<<<(n + 255) / 256, 256, 0, stream>>>(Wih0, Whh0, wc0, IN0); }
  { int n = 1024 * (HID + HID); pack_weights<<<(n + 255) / 256, 256, 0, stream>>>(Wih1, Whh1, wc1, HID); }

  const size_t shm0 = (size_t)(256 * (IN0 + HID + 8) + 32 * (IN0 + 8) + 32 * (HID + 8)) * 2;
  const size_t shm1 = (size_t)(256 * (HID + HID + 8) + 32 * (HID + 8) + 32 * (HID + 8)) * 2;
  (void)hipFuncSetAttribute(reinterpret_cast<const void*>(&lstm_layer_kernel<IN0>),
                            hipFuncAttributeMaxDynamicSharedMemorySize, (int)shm0);
  (void)hipFuncSetAttribute(reinterpret_cast<const void*>(&lstm_layer_kernel<HID>),
                            hipFuncAttributeMaxDynamicSharedMemorySize, (int)shm1);

  lstm_layer_kernel<IN0><<<32, 256, shm0, stream>>>(xbf,   hseq0, wc0, bih0, bhh0, counters);
  lstm_layer_kernel<HID><<<32, 256, shm1, stream>>>(hseq0, hseq1, wc1, bih1, bhh1, counters);
  fc_kernel<<<B_SZ, 128, 0, stream>>>(hseq1, fcw, fcb, out);
}
